// Model_46196668236069
// MI455X (gfx1250) — matrix operand order repaired and checked by reading, not compiled, not run on hardware
//
#include <hip/hip_runtime.h>

#define N_NODES   50000
#define N_EDGES   1600000
#define N_FEAT    512
#define N_HID     128
#define N_CLASSES 2
#define NUM_GRAPHS 128

typedef __attribute__((ext_vector_type(16))) _Float16 v16h;
typedef __attribute__((ext_vector_type(8)))  _Float16 v8h;
typedef __attribute__((ext_vector_type(8)))  float    v8f;

union V16HU { v16h v; v8h h[2]; };

// ---------------------------------------------------------------------------
// K0: zero/seed workspace. deg starts at 1.0 (self loop), acc and pooled at 0.
// grid covers exactly N_NODES*N_HID = 6,400,000 elements.
// ---------------------------------------------------------------------------
__global__ void k_init(float* __restrict__ deg, float* __restrict__ acc,
                       float* __restrict__ pooled) {
  int i = blockIdx.x * blockDim.x + threadIdx.x;
  acc[i] = 0.0f;                         // grid sized exactly to N_NODES*N_HID
  if (i < N_NODES)            deg[i] = 1.0f;
  if (i < NUM_GRAPHS * N_HID) pooled[i] = 0.0f;
}

// ---------------------------------------------------------------------------
// K1: degree of each destination (float add of 1.0 is exact, order-free).
// ---------------------------------------------------------------------------
__global__ void k_degree(const int* __restrict__ ei, float* __restrict__ deg) {
  int e = blockIdx.x * blockDim.x + threadIdx.x;
  if (e < N_EDGES) atomicAdd(&deg[ei[N_EDGES + e]], 1.0f);
}

// ---------------------------------------------------------------------------
// K2: dinv = rsqrt(deg) in place (deg >= 1 always, so no zero guard needed).
// ---------------------------------------------------------------------------
__global__ void k_dinv(float* __restrict__ deg) {
  int i = blockIdx.x * blockDim.x + threadIdx.x;
  if (i < N_NODES) deg[i] = rsqrtf(deg[i]);
}

// ---------------------------------------------------------------------------
// K3: hs = (x @ W1) * dinv[row], via v_wmma_f32_16x16x32_f16.
// Block: 256 threads = 8 waves; M-tile 128 (16 rows/wave), N-tile = full 128.
// W1 K-slice (32 x 128) is staged transposed into LDS as f16 each K-step.
// N_NODES % 16 == 0, so each wave's 16-row strip is uniformly in- or out-of-
// bounds: the epilogue guard is one scalar branch, no per-lane exec masking.
// ---------------------------------------------------------------------------
#define KSTEP      32
#define LDS_STRIDE 40   // f16 units per row: 80B stride -> 16B aligned, bank-conflict-free

__global__ __launch_bounds__(256) void k_gemm(const float* __restrict__ x,
                                              const float* __restrict__ W1,
                                              const float* __restrict__ dinv,
                                              float* __restrict__ hs) {
  __shared__ _Float16 w1t[N_HID * LDS_STRIDE];   // 10 KB

  const int t     = threadIdx.x;
  const int wave  = t >> 5;
  const int lane  = t & 31;
  const int lhalf = lane >> 4;   // 0: lanes 0-15, 1: lanes 16-31
  const int l16   = lane & 15;

  const int rowBase   = blockIdx.x * 128 + wave * 16;
  const bool waveLive = (rowBase < N_NODES);     // uniform across the wave
  const int arow      = waveLive ? (rowBase + l16) : 0;   // dead waves read row 0
  const float* xrow   = x + (size_t)arow * N_FEAT;

  v8f acc[8];
#pragma unroll
  for (int i = 0; i < 8; ++i) acc[i] = (v8f){};

  for (int k0 = 0; k0 < N_FEAT; k0 += KSTEP) {
    __syncthreads();
    // Stage W1[k0..k0+31][0..127] -> w1t[n][kk] (f16), coalesced global reads.
#pragma unroll
    for (int i = 0; i < 16; ++i) {
      int idx = i * 256 + t;           // 0..4095
      int kk  = idx >> 7;              // 0..31
      int n   = idx & 127;
      w1t[n * LDS_STRIDE + kk] = (_Float16)W1[(k0 + kk) * N_HID + n];
    }
    __syncthreads();

    // A fragment (16x32 f16): lanes 0-15 hold K {0..7,16..23}, lanes 16-31
    // hold K {8..15,24..31}; row M = lane%16.  (ISA 7.12.2)
    V16HU a;
    {
      const float* p0 = xrow + k0 + lhalf * 8;        // K chunk 0..7 / 8..15
      const float* p1 = xrow + k0 + 16 + lhalf * 8;   // K chunk 16..23 / 24..31
      float4 c0 = ((const float4*)p0)[0];
      float4 c1 = ((const float4*)p0)[1];
      float4 c2 = ((const float4*)p1)[0];
      float4 c3 = ((const float4*)p1)[1];
      a.v[0]=(_Float16)c0.x;  a.v[1]=(_Float16)c0.y;
      a.v[2]=(_Float16)c0.z;  a.v[3]=(_Float16)c0.w;
      a.v[4]=(_Float16)c1.x;  a.v[5]=(_Float16)c1.y;
      a.v[6]=(_Float16)c1.z;  a.v[7]=(_Float16)c1.w;
      a.v[8]=(_Float16)c2.x;  a.v[9]=(_Float16)c2.y;
      a.v[10]=(_Float16)c2.z; a.v[11]=(_Float16)c2.w;
      a.v[12]=(_Float16)c3.x; a.v[13]=(_Float16)c3.y;
      a.v[14]=(_Float16)c3.z; a.v[15]=(_Float16)c3.w;
    }

    // B fragments (32x16 f16): lane holds column N = nt*16 + lane%16,
    // K = k0 + (lane/16)*8 + {0..7, 16..23} (matches A).
#pragma unroll
    for (int nt = 0; nt < 8; ++nt) {
      V16HU b;
      const _Float16* bp = &w1t[(nt * 16 + l16) * LDS_STRIDE + lhalf * 8];
      b.h[0] = *(const v8h*)(bp);
      b.h[1] = *(const v8h*)(bp + 16);
      acc[nt] = __builtin_amdgcn_wmma_f32_16x16x32_f16(
          false, a.v, false, b.v, (short)0, acc[nt], false, false);
    }
  }

  // Epilogue: uniform per-wave guard, no per-lane bounds checks.
  if (!waveLive) return;

  // C layout: VGPR r -> row M = r + (lane>=16 ? 8 : 0), col N = lane%16.
  const int mBase = rowBase + lhalf * 8;
  float dv[8];
#pragma unroll
  for (int r = 0; r < 8; ++r) dv[r] = dinv[mBase + r];
#pragma unroll
  for (int nt = 0; nt < 8; ++nt) {
#pragma unroll
    for (int r = 0; r < 8; ++r) {
      hs[(size_t)(mBase + r) * N_HID + nt * 16 + l16] = acc[nt][r] * dv[r];
    }
  }
}

// ---------------------------------------------------------------------------
// K4: acc[dst] += hs[src] over all edges. 32 threads/edge, float4 per thread.
// hs (25.6MB) and acc (25.6MB) both fit in the 192MB L2 -> atomics stay on-chip.
// ---------------------------------------------------------------------------
__global__ void k_scatter(const int* __restrict__ ei,
                          const float* __restrict__ hs,
                          float* __restrict__ acc) {
  unsigned u = blockIdx.x * blockDim.x + threadIdx.x;
  unsigned e = u >> 5;
  if (e >= N_EDGES) return;
  int c = (u & 31) * 4;
  int s = ei[e];
  int d = ei[N_EDGES + e];
  const float4 m = *(const float4*)(hs + (size_t)s * N_HID + c);
  float* ap = acc + (size_t)d * N_HID + c;
  atomicAdd(ap + 0, m.x);
  atomicAdd(ap + 1, m.y);
  atomicAdd(ap + 2, m.z);
  atomicAdd(ap + 3, m.w);
}

// ---------------------------------------------------------------------------
// K5: out = relu(dinv*(acc+hs) + b1); segment-max pool via int atomicMax
// (valid since relu output >= 0 and pooled is 0-initialized). Each thread
// handles 8 consecutive nodes of one feature, flushing on batch-id change
// (segments are ~390 nodes long, so ~1 atomic per 8 nodes).
// ---------------------------------------------------------------------------
__global__ void k_pool(const float* __restrict__ dinv,
                       const float* __restrict__ hs,
                       const float* __restrict__ acc,
                       const float* __restrict__ b1,
                       const int* __restrict__ batch,
                       int* __restrict__ pooled) {
  int u = blockIdx.x * blockDim.x + threadIdx.x;
  int f = u & (N_HID - 1);
  int v0 = (u >> 7) * 8;
  if (v0 >= N_NODES) return;
  float bias = b1[f];
  int curB = -1; float curM = 0.0f;
  for (int j = 0; j < 8; ++j) {
    int v = v0 + j;
    if (v >= N_NODES) break;
    float val = dinv[v] * (acc[(size_t)v * N_HID + f] + hs[(size_t)v * N_HID + f]) + bias;
    val = fmaxf(val, 0.0f);
    int b = batch[v];
    if (b != curB) {
      if (curB >= 0) atomicMax(&pooled[curB * N_HID + f], __float_as_int(curM));
      curB = b; curM = val;
    } else {
      curM = fmaxf(curM, val);
    }
  }
  if (curB >= 0) atomicMax(&pooled[curB * N_HID + f], __float_as_int(curM));
}

// ---------------------------------------------------------------------------
// K6: logits = pooled @ W2 + b2; log_softmax over 2 classes. 128 threads.
// ---------------------------------------------------------------------------
__global__ void k_head(const float* __restrict__ pooled,
                       const float* __restrict__ W2,
                       const float* __restrict__ b2,
                       float* __restrict__ out) {
  int g = threadIdx.x;
  if (g >= NUM_GRAPHS) return;
  float l0 = b2[0], l1 = b2[1];
  for (int k = 0; k < N_HID; ++k) {
    float p = pooled[g * N_HID + k];
    l0 += p * W2[k * N_CLASSES + 0];
    l1 += p * W2[k * N_CLASSES + 1];
  }
  float m = fmaxf(l0, l1);
  float lse = m + logf(expf(l0 - m) + expf(l1 - m));
  out[g * 2 + 0] = l0 - lse;
  out[g * 2 + 1] = l1 - lse;
}

// ---------------------------------------------------------------------------
extern "C" void kernel_launch(void* const* d_in, const int* in_sizes, int n_in,
                              void* d_out, int out_size, void* d_ws, size_t ws_size,
                              hipStream_t stream) {
  const float* x   = (const float*)d_in[0];
  const float* W1  = (const float*)d_in[1];
  const float* b1  = (const float*)d_in[2];
  const float* W2  = (const float*)d_in[3];
  const float* b2  = (const float*)d_in[4];
  const int*   ei  = (const int*)d_in[5];    // [2, E] int32 per harness
  const int*   bat = (const int*)d_in[6];    // [N] int32
  float* out = (float*)d_out;

  // Workspace layout (floats, 16-float aligned): ~51.5 MB total
  float* deg    = (float*)d_ws;          // 50048  (deg, then dinv in place)
  float* pooled = deg + 50048;           // 16384
  float* hs     = pooled + 16384;        // 6,400,000
  float* acc    = hs + 6400000;          // 6,400,000

  k_init   <<<(N_NODES * N_HID) / 256, 256, 0, stream>>>(deg, acc, pooled);
  k_degree <<<(N_EDGES + 255) / 256,   256, 0, stream>>>(ei, deg);
  k_dinv   <<<(N_NODES + 255) / 256,   256, 0, stream>>>(deg);
  k_gemm   <<<(N_NODES + 127) / 128,   256, 0, stream>>>(x, W1, deg, hs);
  k_scatter<<<(N_EDGES * 32) / 256,    256, 0, stream>>>(ei, hs, acc);
  {
    int groups = (N_NODES + 7) / 8;
    k_pool <<<(groups * N_HID + 255) / 256, 256, 0, stream>>>(deg, hs, acc, b1, bat,
                                                              (int*)pooled);
  }
  k_head   <<<1, 128, 0, stream>>>(pooled, W2, b2, out);
}
